// DiagSSMLayer_87041807220866
// MI455X (gfx1250) — matrix operand order repaired and checked by reading, not compiled, not run on hardware
//
#include <hip/hip_runtime.h>

typedef __attribute__((ext_vector_type(16))) __bf16 v16bf;
typedef __attribute__((ext_vector_type(8)))  float  v8f;

constexpr int Bsz = 8, T = 4096, D = 1024, H = 1024;
constexpr int M = Bsz * T;               // 32768 GEMM rows
constexpr int CHUNKS = 64, CLEN = 64;    // CHUNKS * CLEN == T

#if __has_builtin(__builtin_amdgcn_global_load_async_to_lds_b128) && \
    __has_builtin(__builtin_amdgcn_s_wait_asynccnt)
#define HAVE_ASYNC_LDS 1
typedef int int4v __attribute__((ext_vector_type(4)));
typedef __attribute__((address_space(1))) int4v as1_int4;
typedef __attribute__((address_space(3))) int4v as3_int4;
#else
#define HAVE_ASYNC_LDS 0
#endif

// ---------------------------------------------------------------------------
// Kernel 0: transpose W [H][D] -> Wt [D][H], split fp32 into bf16 hi + lo.
// ---------------------------------------------------------------------------
__global__ void wt_split(const float* __restrict__ W,
                         __bf16* __restrict__ wh, __bf16* __restrict__ wl) {
    int idx = blockIdx.x * 256 + threadIdx.x;   // over H*D, coalesced read
    int h = idx / D, d = idx % D;
    float w = W[idx];
    __bf16 hi = (__bf16)w;
    __bf16 lo = (__bf16)(w - (float)hi);
    wh[(size_t)d * H + h] = hi;
    wl[(size_t)d * H + h] = lo;
}

// ---------------------------------------------------------------------------
// Kernel 1: h_in = u @ W^T via bf16x3 split-precision WMMA.
// Block = 256 thr = 8 waves (4m x 2n); block tile 128M x 64N.
// Each wave computes 2x2 = 4 16x16 tiles (12 WMMAs per K-step).
// B panel staged in LDS via async global->LDS copies (ASYNCcnt-tracked).
// ---------------------------------------------------------------------------
__global__ void __launch_bounds__(256)
gemm_bf16x3(const float* __restrict__ u, const __bf16* __restrict__ wh,
            const __bf16* __restrict__ wl, float* __restrict__ hin) {
    constexpr int BPITCH = 72;                       // 64 + 8 pad (bank stagger)
    __shared__ __bf16 sBh[32][BPITCH];
    __shared__ __bf16 sBl[32][BPITCH];

    const int lane = threadIdx.x & 31;
    const int wv   = threadIdx.x >> 5;               // 0..7
    const int mw   = wv >> 1;                        // 0..3
    const int nw   = wv & 1;                         // 0..1
    const int half = lane >> 4;                      // 0/1
    const int l16  = lane & 15;

    const int mBase = blockIdx.x * 128 + mw * 32;    // wave's 32 M rows
    const int nBase = blockIdx.y * 64 + nw * 32;     // wave's 32 N cols
    const int nLoc  = nw * 32;                       // within LDS panel

    // async-copy mapping: thread -> (row 0..31, 16B segment 0..7) of B panel
    const int cr  = threadIdx.x >> 3;                // 0..31  (K row)
    const int cs  = (threadIdx.x & 7) * 8;           // col in bf16 elems
    const size_t gb = (size_t)blockIdx.y * 64 + cs;  // N offset in Wt rows

    const float* __restrict__ arow0 = u + (size_t)(mBase + l16) * D;
    const float* __restrict__ arow1 = u + (size_t)(mBase + 16 + l16) * D;
    const int krow = l16 + (half << 4);              // B fragment K-row 0..31

    v8f acc[2][2] = {};

    for (int k0 = 0; k0 < D; k0 += 32) {
        // ---- stage B panel (hi+lo) into LDS -------------------------------
        const __bf16* gh = wh + (size_t)(k0 + cr) * H + gb;
        const __bf16* gl = wl + (size_t)(k0 + cr) * H + gb;
#if HAVE_ASYNC_LDS
        __builtin_amdgcn_global_load_async_to_lds_b128(
            (as1_int4*)(void*)gh, (as3_int4*)&sBh[cr][cs], 0, 0);
        __builtin_amdgcn_global_load_async_to_lds_b128(
            (as1_int4*)(void*)gl, (as3_int4*)&sBl[cr][cs], 0, 0);
        __builtin_amdgcn_s_wait_asynccnt(0);
        __syncthreads();
#else
        *(uint4*)&sBh[cr][cs] = *(const uint4*)gh;
        *(uint4*)&sBl[cr][cs] = *(const uint4*)gl;
        __syncthreads();
#endif

        // ---- A fragments: load fp32, split into bf16 hi/lo ----------------
        if (k0 + 32 < D) __builtin_prefetch(arow0 + k0 + 32, 0, 1);
        v16bf ahi[2], alo[2];
#pragma unroll
        for (int mt = 0; mt < 2; ++mt) {
            const float* __restrict__ ar = mt ? arow1 : arow0;
            const float4 f0 = *(const float4*)(ar + k0 + half * 8);
            const float4 f1 = *(const float4*)(ar + k0 + half * 8 + 4);
            const float4 f2 = *(const float4*)(ar + k0 + 16 + half * 8);
            const float4 f3 = *(const float4*)(ar + k0 + 16 + half * 8 + 4);
            float af[16] = {f0.x, f0.y, f0.z, f0.w, f1.x, f1.y, f1.z, f1.w,
                            f2.x, f2.y, f2.z, f2.w, f3.x, f3.y, f3.z, f3.w};
#pragma unroll
            for (int i = 0; i < 16; ++i) {
                __bf16 hi = (__bf16)af[i];
                ahi[mt][i] = hi;
                alo[mt][i] = (__bf16)(af[i] - (float)hi);
            }
        }

        // ---- B fragments from LDS -----------------------------------------
        v16bf bhi[2], blo[2];
#pragma unroll
        for (int nt = 0; nt < 2; ++nt) {
#pragma unroll
            for (int s = 0; s < 16; ++s) {
                const int kk = (s < 8) ? (half * 8 + s) : (16 + half * 8 + (s - 8));
                bhi[nt][s] = sBh[kk][nLoc + nt * 16 + l16];
                blo[nt][s] = sBl[kk][nLoc + nt * 16 + l16];
            }
        }

        // ---- 2x2 tiles x bf16x3 = 12 WMMAs --------------------------------
#pragma unroll
        for (int mt = 0; mt < 2; ++mt)
#pragma unroll
            for (int nt = 0; nt < 2; ++nt) {
                acc[mt][nt] = __builtin_amdgcn_wmma_f32_16x16x32_bf16(
                    false, ahi[mt], false, bhi[nt], (short)0, acc[mt][nt], false, false);
                acc[mt][nt] = __builtin_amdgcn_wmma_f32_16x16x32_bf16(
                    false, ahi[mt], false, blo[nt], (short)0, acc[mt][nt], false, false);
                acc[mt][nt] = __builtin_amdgcn_wmma_f32_16x16x32_bf16(
                    false, alo[mt], false, bhi[nt], (short)0, acc[mt][nt], false, false);
            }
        __syncthreads();   // protect LDS panel before next stage overwrites
    }

    // ---- store C: VGPR v -> M = v + 8*half (+tile), N = lane%16 -----------
#pragma unroll
    for (int mt = 0; mt < 2; ++mt) {
        const int crow0 = mBase + mt * 16 + half * 8;
#pragma unroll
        for (int nt = 0; nt < 2; ++nt) {
            const int ccol = nBase + nt * 16 + l16;
#pragma unroll
            for (int v = 0; v < 8; ++v)
                hin[(size_t)(crow0 + v) * H + ccol] = acc[mt][nt][v];
        }
    }
}

// ---------------------------------------------------------------------------
// Blocked scan: x_t = a*x_{t-1} + b*h_t.  Per chunk: x_{t0+k} = s_k + a^{k+1}*X0.
// Phase 1: per-chunk local carries s_{L-1} (read-only over h_in).
// ---------------------------------------------------------------------------
__global__ void scan_phase1(const float* __restrict__ buf,
                            const float* __restrict__ a, const float* __restrict__ b,
                            float* __restrict__ carry) {
    int tid   = blockIdx.x * 256 + threadIdx.x;   // B*CHUNKS*H threads
    int h     = tid % H;
    int chunk = (tid / H) % CHUNKS;
    int bb    = tid / (H * CHUNKS);
    float ah = a[h], bh = b[h];
    size_t base = ((size_t)(bb * T + chunk * CLEN)) * H + h;
    float s = 0.f;
    for (int k = 0; k < CLEN; ++k)
        s = ah * s + bh * buf[base + (size_t)k * H];
    carry[((size_t)bb * CHUNKS + chunk) * H + h] = s;
}

// Phase 2: sequential combine across chunks per (b,h); emits chunk-start states
// X0 and the final state xT.
__global__ void scan_phase2(const float* __restrict__ carry,
                            const float* __restrict__ a,
                            float* __restrict__ X0, float* __restrict__ xT) {
    int tid = blockIdx.x * 256 + threadIdx.x;     // B*H threads
    int h = tid % H, bb = tid / H;
    float ah = a[h];
    float aL = 1.f;
#pragma unroll
    for (int i = 0; i < CLEN; ++i) aL *= ah;      // a^CLEN
    float state = 0.f;
    for (int c = 0; c < CHUNKS; ++c) {
        size_t i = ((size_t)bb * CHUNKS + c) * H + h;
        X0[i] = state;
        state = aL * state + carry[i];
    }
    xT[(size_t)bb * H + h] = state;
}

// Phase 3: recompute local recurrence, apply chunk-start correction, emit y
// in place over h_in:  y_k = c*(s_k + a^{k+1}*X0) + out_bias.
__global__ void scan_phase3(float* __restrict__ buf,
                            const float* __restrict__ a, const float* __restrict__ b,
                            const float* __restrict__ c, const float* __restrict__ ob,
                            const float* __restrict__ X0) {
    int tid   = blockIdx.x * 256 + threadIdx.x;
    int h     = tid % H;
    int chunk = (tid / H) % CHUNKS;
    int bb    = tid / (H * CHUNKS);
    float ah = a[h], bh = b[h], ch = c[h], obh = ob[h];
    float x0 = X0[((size_t)bb * CHUNKS + chunk) * H + h];
    size_t base = ((size_t)(bb * T + chunk * CLEN)) * H + h;
    float s = 0.f, p = ah;                        // p = a^{k+1}
    for (int k = 0; k < CLEN; ++k) {
        size_t idx = base + (size_t)k * H;
        s = ah * s + bh * buf[idx];
        buf[idx] = ch * (s + p * x0) + obh;
        p *= ah;
    }
}

// ---------------------------------------------------------------------------
extern "C" void kernel_launch(void* const* d_in, const int* in_sizes, int n_in,
                              void* d_out, int out_size, void* d_ws, size_t ws_size,
                              hipStream_t stream) {
    const float* u  = (const float*)d_in[0];   // [B,T,D]
    const float* W  = (const float*)d_in[1];   // [H,D]
    const float* a  = (const float*)d_in[2];   // [H]
    const float* b  = (const float*)d_in[3];   // [H]
    const float* c  = (const float*)d_in[4];   // [H]
    const float* ob = (const float*)d_in[5];   // [H]

    float* y  = (float*)d_out;                 // [B,T,H]
    float* xT = y + (size_t)M * H;             // [B,H]

    // ws layout: Wt_hi (2MB bf16) | Wt_lo (2MB) | carry (2MB f32) | X0 (2MB)
    char*   ws    = (char*)d_ws;
    __bf16* wh    = (__bf16*)ws;
    __bf16* wl    = (__bf16*)(ws + (size_t)D * H * sizeof(__bf16));
    float*  carry = (float*)(ws + 2 * (size_t)D * H * sizeof(__bf16));
    float*  X0    = carry + (size_t)Bsz * CHUNKS * H;

    wt_split<<<(H * D) / 256, 256, 0, stream>>>(W, wh, wl);

    dim3 ggrid(M / 128, H / 64);
    gemm_bf16x3<<<ggrid, 256, 0, stream>>>(u, wh, wl, y);

    scan_phase1<<<(Bsz * CHUNKS * H) / 256, 256, 0, stream>>>(y, a, b, carry);
    scan_phase2<<<(Bsz * H) / 256, 256, 0, stream>>>(carry, a, X0, xT);
    scan_phase3<<<(Bsz * CHUNKS * H) / 256, 256, 0, stream>>>(y, a, b, c, ob, X0);
}
